// BFR_28089086116615
// MI455X (gfx1250) — matrix operand order repaired and checked by reading, not compiled, not run on hardware
//
#include <hip/hip_runtime.h>
#include <hip/hip_bf16.h>

#define B_  2
#define G_  4096
#define NIN 8
#define H_  32
#define NOUT 32
#define JSPLIT 8

typedef __attribute__((ext_vector_type(16))) __bf16 v16bf;
typedef __attribute__((ext_vector_type(8)))  float  v8f;
typedef __attribute__((ext_vector_type(4)))  float  v4f;
typedef unsigned int uint32;
typedef __attribute__((ext_vector_type(8)))  uint32 v8ui;
typedef __attribute__((ext_vector_type(16))) unsigned short v16us;

static __device__ __forceinline__ unsigned short f2bf(float f) {
  uint32 u = __builtin_bit_cast(uint32, f);
  uint32 r = (u + 0x7FFFu + ((u >> 16) & 1u)) >> 16;
  return (unsigned short)r;
}
static __device__ __forceinline__ float eluf(float x) {
  return x > 0.f ? x : expm1f(x);
}

// ---------------------------------------------------------------- zero
__global__ __launch_bounds__(256) void zero_kernel(float* __restrict__ p, int n) {
  int i = blockIdx.x * 256 + threadIdx.x;
  if (i < n) p[i] = 0.f;
}

// ---------------------------------------------------------------- infer + edge projections
// h = elu(x @ W_infer + b_infer); also bf16 copy and s_src/s_dst = h @ W_e[:H]/[H:]
__global__ __launch_bounds__(256) void infer_proj_kernel(
    const float* __restrict__ x, const float* __restrict__ Wi, const float* __restrict__ bi,
    const float* __restrict__ We,
    float* __restrict__ h, unsigned short* __restrict__ hbf,
    float* __restrict__ ssrc, float* __restrict__ sdst) {
  int tid = blockIdx.x * 256 + threadIdx.x;
  int row = tid >> 5;            // b*G + g  (8192 rows total)
  int n   = tid & 31;
  const float* xr = x + (size_t)row * NIN;
  float acc = bi[n];
#pragma unroll
  for (int k = 0; k < NIN; ++k) acc += xr[k] * Wi[k * H_ + n];
  float hv = eluf(acc);
  h[(size_t)row * H_ + n]   = hv;
  hbf[(size_t)row * H_ + n] = f2bf(hv);
  float ps = hv * We[n];
  float pd = hv * We[H_ + n];
#pragma unroll
  for (int off = 16; off; off >>= 1) {
    ps += __shfl_xor(ps, off, 32);
    pd += __shfl_xor(pd, off, 32);
  }
  if (n == 0) { ssrc[row] = ps; sdst[row] = pd; }
}

// ---------------------------------------------------------------- pre-swizzle h into WMMA-B layout
// swz[b][c][sel][l][e] = hbf[b][c*32 + (l>>4)*8 + e + (e&8)][(l&15) + sel*16]
// i.e. the exact bf16 B-matrix (32x16) VGPR image for chunk c, so the mp kernel
// can load each lane's 16 B elements as one aligned 32-byte global load.
__global__ __launch_bounds__(256) void swz_kernel(
    const unsigned short* __restrict__ hbf,   // [B][G][H]
    unsigned short* __restrict__ swz) {       // [B][128][2][32][16]
  int tid = blockIdx.x * 256 + threadIdx.x;   // 0..16383 = b*8192 + c*64 + sel*32 + l
  int l   = tid & 31;
  int sel = (tid >> 5) & 1;
  int c   = (tid >> 6) & 127;
  int b   = tid >> 13;
  int col = (l & 15) + sel * 16;
  size_t rbase = ((size_t)b * G_ + c * 32 + ((l >> 4) << 3)) * H_ + col;
  v16us v;
#pragma unroll
  for (int e = 0; e < 16; ++e) v[e] = hbf[rbase + (size_t)(e + (e & 8)) * H_];
  *(v16us*)(swz + (size_t)tid * 16) = v;      // out offset == tid*16 by construction
}

// ---------------------------------------------------------------- heavy WMMA message passing
// acc[b,i,:] += sum_j e[b,i,j] * h[b,j,:]   ;   rowsum[b,i] += sum_j e[b,i,j]
// e built on the fly: sigmoid(sdst_i + ssrc_j + be) * (adj==0 ? coef : adj)
// No LDS: per-lane A/ssrc elements are two contiguous runs of 8 in global memory,
// B comes pre-swizzled from swz_kernel.
__global__ __launch_bounds__(32) void mp_wmma_kernel(
    const unsigned short* __restrict__ swz,   // [B][128][2][32][16] bf16 bits
    const float* __restrict__ ssrc,           // [B][G]
    const float* __restrict__ sdst,           // [B][G]
    const float* __restrict__ adj,            // [G][G]
    const float* __restrict__ be,             // [1]
    float coef,
    float* __restrict__ acc,                  // [B][G][H]
    float* __restrict__ rowsum) {             // [B][G]
  int wid    = blockIdx.x;                    // 0 .. B*(G/16)*JSPLIT - 1
  int jslice = wid % JSPLIT;
  int itile  = (wid / JSPLIT) % (G_ / 16);
  int b      = wid / (JSPLIT * (G_ / 16));
  int i0     = itile * 16;
  int lane   = threadIdx.x;
  int m      = lane & 15;
  int kh8    = (lane >> 4) << 3;

  const float dd = sdst[(size_t)b * G_ + i0 + m] + be[0];
  // fold the log2(e) prescale for the fast exp2-based sigmoid:
  // sigmoid(lg) = 1 / (1 + exp2(-lg * log2e)) ; -lg*log2e = ddn - ss*log2e
  const float LOG2E = 1.4426950408889634f;
  const float ddn = -dd * LOG2E;
  const float* adjRow = adj + (size_t)(i0 + m) * G_;
  const float* sB = ssrc + (size_t)b * G_;
  const unsigned short* swzB = swz + (size_t)b * (128 * 1024) + (size_t)lane * 16;

  v8f c0 = {};
  v8f c1 = {};
  float rs = 0.f;

  for (int j0 = jslice * 32; j0 < G_; j0 += JSPLIT * 32) {
    int j0n = j0 + JSPLIT * 32;
    if (j0n < G_) __builtin_prefetch(adjRow + j0n, 0, 3);   // global_prefetch_b8

    // A-operand adjacency + ssrc: per-lane K-set is [kh8, kh8+8) U [16+kh8, 16+kh8+8)
    v4f Aq[4], Sq[4];
    Aq[0] = *(const v4f*)(adjRow + j0 + kh8);
    Aq[1] = *(const v4f*)(adjRow + j0 + kh8 + 4);
    Aq[2] = *(const v4f*)(adjRow + j0 + 16 + kh8);
    Aq[3] = *(const v4f*)(adjRow + j0 + 16 + kh8 + 4);
    Sq[0] = *(const v4f*)(sB + j0 + kh8);
    Sq[1] = *(const v4f*)(sB + j0 + kh8 + 4);
    Sq[2] = *(const v4f*)(sB + j0 + 16 + kh8);
    Sq[3] = *(const v4f*)(sB + j0 + 16 + kh8 + 4);

    // B-operand: one aligned 32B load per matrix (pre-swizzled, L2-resident)
    const unsigned short* sw = swzB + (size_t)(j0 >> 5) * 1024;
    v16bf b0 = __builtin_bit_cast(v16bf, *(const v8ui*)(sw));
    v16bf b1 = __builtin_bit_cast(v16bf, *(const v8ui*)(sw + 512));

    v16bf a;
    float lrs = 0.f;
#pragma unroll
    for (int e = 0; e < 16; ++e) {
      float w = Aq[e >> 2][e & 3];
      w = (w == 0.f) ? coef : w;
      // fast sigmoid: v_fma + v_exp_f32 (trans, co-executes with WMMA) + v_add + v_rcp + v_mul
      float t   = __builtin_amdgcn_exp2f(__builtin_fmaf(Sq[e >> 2][e & 3], -LOG2E, ddn));
      float eij = w * __builtin_amdgcn_rcpf(1.f + t);
      a[e] = (__bf16)eij;
      lrs += eij;
    }
    rs += lrs;
    c0 = __builtin_amdgcn_wmma_f32_16x16x32_bf16(false, a, false, b0, (short)0, c0, false, false);
    c1 = __builtin_amdgcn_wmma_f32_16x16x32_bf16(false, a, false, b1, (short)0, c1, false, false);
  }

  // C layout: VGPR r holds M = r + 8*kh, N = m. Resolve J-split with f32 atomics.
  int kh = lane >> 4;
#pragma unroll
  for (int r = 0; r < 8; ++r) {
    int row = i0 + r + (kh << 3);
    size_t o = ((size_t)b * G_ + row) * H_ + m;
    __hip_atomic_fetch_add(&acc[o],      c0[r], __ATOMIC_RELAXED, __HIP_MEMORY_SCOPE_AGENT);
    __hip_atomic_fetch_add(&acc[o + 16], c1[r], __ATOMIC_RELAXED, __HIP_MEMORY_SCOPE_AGENT);
  }
  rs += __shfl_xor(rs, 16, 32);
  if (kh == 0)
    __hip_atomic_fetch_add(&rowsum[(size_t)b * G_ + i0 + m], rs,
                           __ATOMIC_RELAXED, __HIP_MEMORY_SCOPE_AGENT);
}

// ---------------------------------------------------------------- node MLP + merge MLP
__global__ __launch_bounds__(256) void post_kernel(
    const float* __restrict__ acc, const float* __restrict__ rowsum,
    const float* __restrict__ h,
    const float* __restrict__ Wn, const float* __restrict__ bn,
    const float* __restrict__ Wm, const float* __restrict__ bm,
    float* __restrict__ out) {
  __shared__ float sRecv[8][32];
  int tid  = threadIdx.x;
  int n    = tid & 31;
  int rloc = tid >> 5;
  int row  = blockIdx.x * 8 + rloc;      // b*G + g
  const float* src = acc + (size_t)row * NOUT;
  const float* hr  = h + (size_t)row * H_;
  float rs = rowsum[row];
  float a = bn[n];
  for (int k = 0; k < H_; ++k) a += src[k] * Wn[k * NOUT + n];
  for (int k = 0; k < H_; ++k) a += (hr[k] * rs) * Wn[(H_ + k) * NOUT + n];
  float recv = eluf(a);
  sRecv[rloc][n] = recv;
  __syncthreads();
  float u = bm[n];
  for (int k = 0; k < NOUT; ++k) u += sRecv[rloc][k] * Wm[k * NOUT + n];
  for (int k = 0; k < H_; ++k)   u += hr[k] * Wm[(NOUT + k) * NOUT + n];
  out[(size_t)row * NOUT + n] = eluf(u);
}

// ---------------------------------------------------------------- BatchNorm per gene + block-2 projections
__global__ __launch_bounds__(256) void bn_proj_kernel(
    const float* __restrict__ hin, const float* __restrict__ gamma, const float* __restrict__ beta,
    const float* __restrict__ We2,
    float* __restrict__ hout, unsigned short* __restrict__ hbf,
    float* __restrict__ ssrc, float* __restrict__ sdst) {
  int tid = blockIdx.x * 256 + threadIdx.x;
  int g = tid >> 5;
  int n = tid & 31;
  float v0 = hin[(size_t)g * NOUT + n];
  float v1 = hin[((size_t)G_ + g) * NOUT + n];
  float s = v0 + v1, sq = v0 * v0 + v1 * v1;
#pragma unroll
  for (int off = 16; off; off >>= 1) {
    s  += __shfl_xor(s, off, 32);
    sq += __shfl_xor(sq, off, 32);
  }
  float mu  = s * (1.f / 64.f);
  float var = sq * (1.f / 64.f) - mu * mu;
  float inv = rsqrtf(var + 1e-5f);
  float gm = gamma[g], bt = beta[g];
  float o0 = (v0 - mu) * inv * gm + bt;
  float o1 = (v1 - mu) * inv * gm + bt;
  hout[(size_t)g * NOUT + n]        = o0;
  hout[((size_t)G_ + g) * NOUT + n] = o1;
  hbf[(size_t)g * NOUT + n]         = f2bf(o0);
  hbf[((size_t)G_ + g) * NOUT + n]  = f2bf(o1);
  float ws = We2[n], wd = We2[H_ + n];
  float p0s = o0 * ws, p0d = o0 * wd, p1s = o1 * ws, p1d = o1 * wd;
#pragma unroll
  for (int off = 16; off; off >>= 1) {
    p0s += __shfl_xor(p0s, off, 32);
    p0d += __shfl_xor(p0d, off, 32);
    p1s += __shfl_xor(p1s, off, 32);
    p1d += __shfl_xor(p1d, off, 32);
  }
  if (n == 0) {
    ssrc[g] = p0s;       sdst[g] = p0d;
    ssrc[G_ + g] = p1s;  sdst[G_ + g] = p1d;
  }
}

// ---------------------------------------------------------------- launcher
extern "C" void kernel_launch(void* const* d_in, const int* in_sizes, int n_in,
                              void* d_out, int out_size, void* d_ws, size_t ws_size,
                              hipStream_t stream) {
  const float* x       = (const float*)d_in[0];
  const float* edges1  = (const float*)d_in[1];
  const float* edges2  = (const float*)d_in[2];
  const float* W_infer = (const float*)d_in[3];
  const float* b_infer = (const float*)d_in[4];
  const float* W_e1    = (const float*)d_in[5];
  const float* b_e1    = (const float*)d_in[6];
  const float* W_e2    = (const float*)d_in[7];
  const float* b_e2    = (const float*)d_in[8];
  const float* W_n1    = (const float*)d_in[9];
  const float* b_n1    = (const float*)d_in[10];
  const float* W_n2    = (const float*)d_in[11];
  const float* b_n2    = (const float*)d_in[12];
  const float* W_m1    = (const float*)d_in[13];
  const float* b_m1    = (const float*)d_in[14];
  const float* W_m2    = (const float*)d_in[15];
  const float* b_m2    = (const float*)d_in[16];
  const float* gamma   = (const float*)d_in[17];
  const float* beta    = (const float*)d_in[18];

  const int rows = B_ * G_;                       // 8192
  char* p = (char*)d_ws;
  float*          h1   = (float*)p;          p += (size_t)rows * H_ * 4;     // 1 MB
  unsigned short* h1bf = (unsigned short*)p; p += (size_t)rows * H_ * 2;     // 0.5 MB
  float*          s1s  = (float*)p;          p += (size_t)rows * 4;
  float*          s1d  = (float*)p;          p += (size_t)rows * 4;
  float*          acc  = (float*)p;          p += (size_t)rows * NOUT * 4;   // 1 MB (rowsum follows, zeroed together)
  float*          rsum = (float*)p;          p += (size_t)rows * 4;
  float*          hmid = (float*)p;          p += (size_t)rows * NOUT * 4;   // 1 MB
  float*          h2   = (float*)p;          p += (size_t)rows * H_ * 4;     // 1 MB
  unsigned short* h2bf = (unsigned short*)p; p += (size_t)rows * H_ * 2;
  float*          s2s  = (float*)p;          p += (size_t)rows * 4;
  float*          s2d  = (float*)p;          p += (size_t)rows * 4;
  unsigned short* swz  = (unsigned short*)p; p += (size_t)B_ * 128 * 1024 * 2; // 0.5 MB, reused by both blocks
  (void)ws_size; (void)in_sizes; (void)n_in; (void)out_size;

  const int nAccZero = rows * NOUT + rows;        // acc + rsum contiguous
  const int mpBlocks = B_ * (G_ / 16) * JSPLIT;   // 4096 waves

  // h = elu(x W + b) ; bf16 copy ; block-1 edge projections
  infer_proj_kernel<<<rows * 32 / 256, 256, 0, stream>>>(
      x, W_infer, b_infer, W_e1, h1, h1bf, s1s, s1d);

  // ---- message-passing block 1
  swz_kernel<<<64, 256, 0, stream>>>(h1bf, swz);
  zero_kernel<<<(nAccZero + 255) / 256, 256, 0, stream>>>(acc, nAccZero);
  mp_wmma_kernel<<<mpBlocks, 32, 0, stream>>>(
      swz, s1s, s1d, edges1, b_e1, 0.005f, acc, rsum);
  post_kernel<<<rows / 8, 256, 0, stream>>>(
      acc, rsum, h1, W_n1, b_n1, W_m1, b_m1, hmid);

  // ---- batchnorm + block-2 projections
  bn_proj_kernel<<<G_ * 32 / 256, 256, 0, stream>>>(
      hmid, gamma, beta, W_e2, h2, h2bf, s2s, s2d);

  // ---- message-passing block 2
  swz_kernel<<<64, 256, 0, stream>>>(h2bf, swz);
  zero_kernel<<<(nAccZero + 255) / 256, 256, 0, stream>>>(acc, nAccZero);
  mp_wmma_kernel<<<mpBlocks, 32, 0, stream>>>(
      swz, s2s, s2d, edges2, b_e2, 5e-5f, acc, rsum);
  post_kernel<<<rows / 8, 256, 0, stream>>>(
      acc, rsum, h2, W_n2, b_n2, W_m2, b_m2, (float*)d_out);
}
